// ViT_Custom_MultiHeadAttention_72249939853841
// MI455X (gfx1250) — matrix operand order repaired and checked by reading, not compiled, not run on hardware
//
#include <hip/hip_runtime.h>
#include <cstdint>

// ---------------------------------------------------------------------------
// ViT custom MHA for MI455X (gfx1250, wave32, WMMA bf16 16x16x32).
// All GEMMs are "NT": both operands stored [rows, K] with K contiguous,
// matching the CDNA5 A/B fragment layouts (CDNA5 ISA §7.12.2).
// Double-buffered LDS with GLOBAL_LOAD_ASYNC_TO_LDS (ASYNCcnt) when the
// toolchain exposes the builtin; synchronous ds_store staging otherwise.
// ---------------------------------------------------------------------------

typedef __attribute__((ext_vector_type(16))) __bf16 v16bf;
typedef __attribute__((ext_vector_type(8)))  float  v8f;
typedef __attribute__((ext_vector_type(4)))  int    v4i;

static constexpr int cB = 32, cP = 256, cF = 768, cH = 12, cN = 192;
static constexpr int HP = cH * cP; // 3072

#define AS1 __attribute__((address_space(1)))
#define AS3 __attribute__((address_space(3)))

#if defined(__HIP_DEVICE_COMPILE__) && __has_builtin(__builtin_amdgcn_global_load_async_to_lds_b128)
#define USE_ASYNC_LDS 1
#else
#define USE_ASYNC_LDS 0
#endif

__device__ __forceinline__ void wait_async_zero() {
#if USE_ASYNC_LDS
#if __has_builtin(__builtin_amdgcn_s_wait_asynccnt)
  __builtin_amdgcn_s_wait_asynccnt(0);
#else
  asm volatile("s_wait_asynccnt 0x0" ::: "memory");
#endif
#endif
}

__device__ __forceinline__ unsigned short f2bf(float f) {
  union { float f; unsigned u; } x; x.f = f;
  unsigned r = x.u + 0x7FFFu + ((x.u >> 16) & 1u); // round-to-nearest-even
  return (unsigned short)(r >> 16);
}

// ---------------- elementwise fp32 -> bf16 ----------------
__global__ __launch_bounds__(256)
void k_cvt(const float* __restrict__ in, unsigned short* __restrict__ out, int n) {
  int i = blockIdx.x * blockDim.x + threadIdx.x;
  if (i < n) out[i] = f2bf(in[i]);
}

// ---------------- [F,N,H] fp32 -> [H,N,F] bf16 (per-head, K=F contiguous) ---
__global__ __launch_bounds__(256)
void k_pack_qkvw(const float* __restrict__ src, unsigned short* __restrict__ dst, int n) {
  int i = blockIdx.x * blockDim.x + threadIdx.x;
  if (i >= n) return;
  int f  = i % cF;
  int nn = (i / cF) % cN;
  int h  = i / (cF * cN);
  dst[i] = f2bf(src[(size_t)(f * cN + nn) * cH + h]);
}

// ---------------- transpose-pack: dst[r*C+c] = bf16(src[c*R+r]) -------------
__global__ __launch_bounds__(256)
void k_pack_tr(const float* __restrict__ src, unsigned short* __restrict__ dst, int R, int C) {
  int i = blockIdx.x * blockDim.x + threadIdx.x;
  if (i >= R * C) return;
  int c = i % C, r = i / C;
  dst[i] = f2bf(src[(size_t)c * R + r]);
}

// ---------------- row softmax over 256 f32 -> bf16 (1 wave / row) ----------
__global__ __launch_bounds__(256)
void k_softmax(const float* __restrict__ S, unsigned short* __restrict__ A) {
  int row  = blockIdx.x * 8 + (threadIdx.x >> 5);
  int lane = threadIdx.x & 31;
  const float4* p = (const float4*)(S + (size_t)row * 256 + lane * 8);
  float4 a = p[0], b = p[1];
  float f[8] = { a.x, a.y, a.z, a.w, b.x, b.y, b.z, b.w };
  float m = f[0];
#pragma unroll
  for (int r = 1; r < 8; ++r) m = fmaxf(m, f[r]);
#pragma unroll
  for (int off = 16; off > 0; off >>= 1) m = fmaxf(m, __shfl_xor(m, off, 32));
  float s = 0.f;
#pragma unroll
  for (int r = 0; r < 8; ++r) { f[r] = __expf(f[r] - m); s += f[r]; }
#pragma unroll
  for (int off = 16; off > 0; off >>= 1) s += __shfl_xor(s, off, 32);
  float inv = 1.f / s;
  union { uint4 u; unsigned short h[8]; } pk;
#pragma unroll
  for (int r = 0; r < 8; ++r) pk.h[r] = f2bf(f[r] * inv);
  *(uint4*)(A + (size_t)row * 256 + lane * 8) = pk.u;
}

// ---------------------------------------------------------------------------
// Generic bf16 NT GEMM: C[M,Nc] = A[M,K] * B[Nc,K]^T  (K contiguous in both).
// Block = 64 threads (2 waves), 64x64 tile; wave tile = 32x64 (2x4 WMMA
// accumulators -> 12 ds_load_b128 per 8 WMMAs). K staged in 64-deep chunks
// through double-buffered padded LDS; next chunk's async loads are issued
// before computing the current one (ASYNCcnt pipeline).
//   STORE: 0 = f32 row-major, 1 = bf16 row-major, 2 = bf16 transposed
//   BIASM: 0 = none, 1 = bias[col*biasColStride], 2 = bias[m*biasLd + col]
// z = blockIdx.z decomposed as (z1 = z/zInner, z2 = z%zInner) for batch/head.
// ---------------------------------------------------------------------------
template<int STORE, int BIASM>
__global__ __launch_bounds__(64)
void gemm_nt(const unsigned short* __restrict__ Abase, int64_t aS1, int64_t aS2,
             const unsigned short* __restrict__ Bbase, int64_t bS1, int64_t bS2,
             void* __restrict__ Obase, int64_t oS1, int64_t oS2,
             const float* __restrict__ biasBase, int64_t biasS1, int64_t biasS2,
             int biasColStride, int biasLd,
             int K, int lda, int ldb, int ldo, int zInner, float scale)
{
  __shared__ unsigned short As[2][64 * 72];
  __shared__ unsigned short Bs[2][64 * 72];

  const int z  = blockIdx.z;
  const int z1 = z / zInner, z2 = z - z1 * zInner;
  const unsigned short* A = Abase + z1 * aS1 + z2 * aS2;
  const unsigned short* B = Bbase + z1 * bS1 + z2 * bS2;
  const float* bias = (BIASM != 0) ? (biasBase + z1 * biasS1 + z2 * biasS2) : nullptr;

  const int rowBase = blockIdx.y * 64;
  const int colBase = blockIdx.x * 64;

  const int t    = threadIdx.x;
  const int wave = t >> 5, lane = t & 31;
  const int wr   = wave * 32;          // wave row offset in tile (2 waves)
  const int mlan = lane & 15, hi = lane >> 4;

  // Stage one 64-deep K chunk of A[64xK] and B[64xK] into LDS buffer `buf`.
  auto stage = [&](int k0, int buf) {
#pragma unroll
    for (int i = 0; i < 8; ++i) {
      int slot = t + i * 64;           // 512 x 16B slots per tile
      int row  = slot >> 3;
      int seg  = (slot & 7) * 8;
      const unsigned short* gA = &A[(size_t)(rowBase + row) * lda + k0 + seg];
      const unsigned short* gB = &B[(size_t)(colBase + row) * ldb + k0 + seg];
#if USE_ASYNC_LDS
      __builtin_amdgcn_global_load_async_to_lds_b128(
          (AS1 v4i*)gA, (AS3 v4i*)&As[buf][row * 72 + seg], 0, 0);
      __builtin_amdgcn_global_load_async_to_lds_b128(
          (AS1 v4i*)gB, (AS3 v4i*)&Bs[buf][row * 72 + seg], 0, 0);
#else
      *(uint4*)&As[buf][row * 72 + seg] = *(const uint4*)gA;
      *(uint4*)&Bs[buf][row * 72 + seg] = *(const uint4*)gB;
#endif
    }
  };

  v8f acc[2][4] = {};
  const int NC = K >> 6;               // all K here are multiples of 64

  stage(0, 0);
  wait_async_zero();
  __syncthreads();

  for (int c = 0; c < NC; ++c) {
    const int cur = c & 1;
    if (c + 1 < NC) stage((c + 1) << 6, cur ^ 1);   // prefetch next chunk

#pragma unroll
    for (int kk = 0; kk < 64; kk += 32) {
      v16bf af[2], bb[4];
#pragma unroll
      for (int i = 0; i < 2; ++i) {
        // A-frag: row m = lane&15; VGPRs 0-3 = k[8*hi .. +7], 4-7 = k[16+8*hi .. +7]
        const unsigned short* p = &As[cur][(wr + i * 16 + mlan) * 72 + kk + hi * 8];
        union { uint4 u[2]; v16bf v; } tmp;
        tmp.u[0] = *(const uint4*)p;
        tmp.u[1] = *(const uint4*)(p + 16);
        af[i] = tmp.v;
      }
#pragma unroll
      for (int j = 0; j < 4; ++j) {
        // B-frag: col n = lane&15; VGPRs 0-3 = k[8*hi .. +7], 4-7 = k[16+8*hi .. +7]
        const unsigned short* p = &Bs[cur][(j * 16 + mlan) * 72 + kk + hi * 8];
        union { uint4 u[2]; v16bf v; } tmp;
        tmp.u[0] = *(const uint4*)p;
        tmp.u[1] = *(const uint4*)(p + 16);
        bb[j] = tmp.v;
      }
#pragma unroll
      for (int i = 0; i < 2; ++i)
#pragma unroll
        for (int j = 0; j < 4; ++j)
          acc[i][j] = __builtin_amdgcn_wmma_f32_16x16x32_bf16(
              false, af[i], false, bb[j], (short)0, acc[i][j], false, false);
    }

    wait_async_zero();   // this iteration's async loads (next chunk) done
    __syncthreads();     // all waves done reading `cur`, next chunk visible
  }

  // Epilogue. C-frag layout: VGPR r -> (M = r + 8*hi, N = lane&15).
  const int gmBase = rowBase + wr + hi * 8;
#pragma unroll
  for (int i = 0; i < 2; ++i) {
#pragma unroll
    for (int j = 0; j < 4; ++j) {
      const int gm0 = gmBase + i * 16;
      const int gc  = colBase + j * 16 + mlan;
      float colb = 0.f;
      if (BIASM == 1) colb = bias[(size_t)gc * biasColStride];
      float vals[8];
#pragma unroll
      for (int r = 0; r < 8; ++r) {
        float v = acc[i][j][r] * scale;
        if (BIASM == 1) v += colb;
        if (BIASM == 2) v += bias[(size_t)(gm0 + r) * biasLd + gc];
        vals[r] = v;
      }
      if (STORE == 0) {
        float* O = (float*)Obase + z1 * oS1 + z2 * oS2;
#pragma unroll
        for (int r = 0; r < 8; ++r) O[(size_t)(gm0 + r) * ldo + gc] = vals[r];
      } else if (STORE == 1) {
        unsigned short* O = (unsigned short*)Obase + z1 * oS1 + z2 * oS2;
#pragma unroll
        for (int r = 0; r < 8; ++r) O[(size_t)(gm0 + r) * ldo + gc] = f2bf(vals[r]);
      } else {
        // Transposed: 8 consecutive output columns -> one packed 16B store.
        unsigned short* O = (unsigned short*)Obase + z1 * oS1 + z2 * oS2;
        union { uint4 u; unsigned short s[8]; } pk;
#pragma unroll
        for (int r = 0; r < 8; ++r) pk.s[r] = f2bf(vals[r]);
        *(uint4*)&O[(size_t)gc * ldo + gm0] = pk.u;
      }
    }
  }
}

// ---------------------------------------------------------------------------
extern "C" void kernel_launch(void* const* d_in, const int* in_sizes, int n_in,
                              void* d_out, int out_size, void* d_ws, size_t ws_size,
                              hipStream_t stream) {
  const float* query   = (const float*)d_in[0];   // [B,P,F]
  const float* value   = (const float*)d_in[1];   // [B,P,F]
  const float* query_w = (const float*)d_in[2];   // [F,N,H]
  const float* key_w   = (const float*)d_in[3];
  const float* value_w = (const float*)d_in[4];
  const float* query_b = (const float*)d_in[5];   // [N,H]
  const float* key_b   = (const float*)d_in[6];
  const float* value_b = (const float*)d_in[7];
  const float* out_w1  = (const float*)d_in[8];   // [H*P, P]
  const float* out_b1  = (const float*)d_in[9];   // [N,P]
  const float* out_w2  = (const float*)d_in[10];  // [N,F]
  const float* out_b2  = (const float*)d_in[11];  // [P,F]
  float*       out     = (float*)d_out;           // [B,P,F]

  char* ws = (char*)d_ws;
  size_t off = 0;
  auto sub = [&](size_t bytes) -> void* {
    void* p = ws + off;
    off = (off + bytes + 255) & ~(size_t)255;
    return p;
  };

  const int BPF = cB * cP * cF;                       // 6,291,456
  unsigned short* Xq  = (unsigned short*)sub((size_t)BPF * 2);            // [B,P,F] bf16
  unsigned short* Xv  = (unsigned short*)sub((size_t)BPF * 2);            // [B,P,F] bf16
  unsigned short* WtQ = (unsigned short*)sub((size_t)cH * cN * cF * 2);   // [H,N,F]
  unsigned short* WtK = (unsigned short*)sub((size_t)cH * cN * cF * 2);
  unsigned short* WtV = (unsigned short*)sub((size_t)cH * cN * cF * 2);
  unsigned short* W1t = (unsigned short*)sub((size_t)cP * HP * 2);        // [P,H*P]
  unsigned short* W2t = (unsigned short*)sub((size_t)cF * cN * 2);        // [F,N]
  unsigned short* Qb  = (unsigned short*)sub((size_t)cB * cH * cP * cN * 2); // [B,H,P,N]
  unsigned short* Kb  = (unsigned short*)sub((size_t)cB * cH * cP * cN * 2); // [B,H,P,N]
  unsigned short* Vt  = (unsigned short*)sub((size_t)cB * cH * cN * cP * 2); // [B,H,N,P]
  float*          Sc  = (float*)sub((size_t)cB * cH * cP * cP * 4);          // [B,H,P,P]
  unsigned short* At  = (unsigned short*)sub((size_t)cB * cH * cP * cP * 2); // [B,H,P,P]
  unsigned short* Ot  = (unsigned short*)sub((size_t)cB * cN * HP * 2);      // [B,N,H*P]
  unsigned short* Y1t = (unsigned short*)sub((size_t)cB * cP * cN * 2);      // [B,P,N]
  (void)ws_size; (void)n_in; (void)in_sizes; (void)out_size;

  const int WQKV = cH * cN * cF; // 1,769,472

  // --- pack / convert ---
  k_cvt<<<(BPF + 255) / 256, 256, 0, stream>>>(query, Xq, BPF);
  k_cvt<<<(BPF + 255) / 256, 256, 0, stream>>>(value, Xv, BPF);
  k_pack_qkvw<<<(WQKV + 255) / 256, 256, 0, stream>>>(query_w, WtQ, WQKV);
  k_pack_qkvw<<<(WQKV + 255) / 256, 256, 0, stream>>>(key_w,   WtK, WQKV);
  k_pack_qkvw<<<(WQKV + 255) / 256, 256, 0, stream>>>(value_w, WtV, WQKV);
  k_pack_tr<<<(cP * HP + 255) / 256, 256, 0, stream>>>(out_w1, W1t, cP, HP);
  k_pack_tr<<<(cF * cN + 255) / 256, 256, 0, stream>>>(out_w2, W2t, cF, cN);

  // --- projections: [P,F] x [N,F]^T per (b,h) ---
  dim3 gProj(cN / 64, cP / 64, cB * cH);
  gemm_nt<1, 1><<<gProj, 64, 0, stream>>>(
      Xq, (int64_t)cP * cF, 0, WtQ, 0, (int64_t)cN * cF,
      Qb, (int64_t)cH * cP * cN, (int64_t)cP * cN,
      query_b, 0, 1, cH, 0, cF, cF, cF, cN, cH, 1.f);
  gemm_nt<1, 1><<<gProj, 64, 0, stream>>>(
      Xq, (int64_t)cP * cF, 0, WtK, 0, (int64_t)cN * cF,
      Kb, (int64_t)cH * cP * cN, (int64_t)cP * cN,
      key_b, 0, 1, cH, 0, cF, cF, cF, cN, cH, 1.f);
  gemm_nt<2, 1><<<gProj, 64, 0, stream>>>(            // V stored transposed [N,P]
      Xv, (int64_t)cP * cF, 0, WtV, 0, (int64_t)cN * cF,
      Vt, (int64_t)cH * cN * cP, (int64_t)cN * cP,
      value_b, 0, 1, cH, 0, cF, cF, cF, cP, cH, 1.f);

  // --- scores = Q K^T / sqrt(F): [P,N] x [P,N]^T per (b,h), f32 out ---
  gemm_nt<0, 0><<<dim3(cP / 64, cP / 64, cB * cH), 64, 0, stream>>>(
      Qb, (int64_t)cH * cP * cN, (int64_t)cP * cN,
      Kb, (int64_t)cH * cP * cN, (int64_t)cP * cN,
      Sc, (int64_t)cH * cP * cP, (int64_t)cP * cP,
      nullptr, 0, 0, 1, 1, cN, cN, cN, cP, cH, 0.03608439182435161f);

  // --- softmax over rows of 256 ---
  k_softmax<<<cB * cH * cP / 8, 256, 0, stream>>>(Sc, At);

  // --- O = attn V: [P,P] x [N,P]^T per (b,h); store O^T at col block h*P ---
  gemm_nt<2, 0><<<dim3(cN / 64, cP / 64, cB * cH), 64, 0, stream>>>(
      At, (int64_t)cH * cP * cP, (int64_t)cP * cP,
      Vt, (int64_t)cH * cN * cP, (int64_t)cN * cP,
      Ot, (int64_t)cN * HP, (int64_t)cP,
      nullptr, 0, 0, 1, 1, cP, cP, cP, HP, cH, 1.f);

  // --- y1 = O_cat W1 + b1: [N,H*P] x [P,H*P]^T per b; store y1^T [P,N] ---
  gemm_nt<2, 2><<<dim3(cP / 64, cN / 64, cB), 64, 0, stream>>>(
      Ot, (int64_t)cN * HP, 0, W1t, 0, 0,
      Y1t, (int64_t)cP * cN, 0,
      out_b1, 0, 0, 1, cP, HP, HP, HP, cN, 1, 1.f);

  // --- y2 = y1^T W2 + b2: [P,N] x [F,N]^T per b; f32 out [B,P,F] ---
  gemm_nt<0, 2><<<dim3(cF / 64, cP / 64, cB), 64, 0, stream>>>(
      Y1t, (int64_t)cP * cN, 0, W2t, 0, 0,
      out, (int64_t)cP * cF, 0,
      out_b2, 0, 0, 1, cF, cN, cN, cN, cF, 1, 1.f);
}
